// GAM_Module_37434934952392
// MI455X (gfx1250) — matrix operand order repaired and checked by reading, not compiled, not run on hardware
//
#include <hip/hip_runtime.h>

// CDNA5 (gfx1250) wave32 WMMA / TDM types
typedef __attribute__((ext_vector_type(16))) _Float16     v16h;
typedef __attribute__((ext_vector_type(8)))  float        v8f;
typedef __attribute__((ext_vector_type(4)))  unsigned int u32x4;
typedef __attribute__((ext_vector_type(8)))  unsigned int u32x8;

#define NN      8192            // C*H*W
#define NBATCH  2
#define BLOCK   128             // 4 waves of 32
#define NWAVES  (BLOCK / 32)
#define ROWS_PB (NWAVES * 16)   // 64 output rows per block

// Rank-1 softmax attention:
//   out[b,i] = gamma * (sum_j v_j e_ij)/(sum_j e_ij) + v_i
//   e_ij = exp2(vi2*v_j - mi2),  vi2 = v_i*log2e,  mi2 = row-max * log2e
__global__ __launch_bounds__(BLOCK, 1)
void gam_rank1_attn(const float* __restrict__ x,
                    const float* __restrict__ gamma,
                    float* __restrict__ out)
{
    __shared__ float sv[NN];          // whole batch vector (32 KB of 320 KB WGP LDS)
    __shared__ float sred_max[BLOCK];
    __shared__ float sred_min[BLOCK];
    __shared__ float sS[ROWS_PB];     // softmax denominators
    __shared__ float sT[ROWS_PB];     // v-weighted numerators

    const int tilesPerBatch = NN / ROWS_PB;       // 128
    const int b    = blockIdx.x / tilesPerBatch;
    const int tile = blockIdx.x % tilesPerBatch;
    const float* v = x + b * NN;
    const int tid  = threadIdx.x;

    // ---- stage v into LDS via the Tensor Data Mover (async DMA), wave 0 issues ----
    if (tid < 32) {
        const unsigned long long ga = (unsigned long long)v;          // global byte addr
        const unsigned int lds_off  = (unsigned int)(unsigned long long)&sv[0];
        u32x4 g0;                                   // D# group 0 (ISA §8.3)
        g0[0] = 0x1u;                               // count=1, user descriptor
        g0[1] = lds_off;                            // lds_addr (bytes)
        g0[2] = (unsigned int)(ga & 0xffffffffull); // global_addr[31:0]
        g0[3] = (unsigned int)((ga >> 32) & 0x1ffffffull) | (2u << 30); // addr[56:32] | type=2
        u32x8 g1;                                   // D# group 1 (ISA §8.4)
        g1[0] = (2u << 16);                         // wg_mask=0, data_size=2 (4 bytes)
        g1[1] = (NN & 0xffffu) << 16;               // tensor_dim0[15:0] = 8192
        g1[2] = (1u << 16);                         // tensor_dim0[31:16]=0, tensor_dim1=1
        g1[3] = (NN & 0xffffu) << 16;               // tile_dim0 = 8192
        g1[4] = 1u;                                 // tile_dim1=1, tile_dim2=0
        g1[5] = (unsigned int)NN;                   // tensor_dim0_stride[31:0]
        g1[6] = 0u;                                 // stride0 hi, stride1 lo
        g1[7] = 0u;                                 // stride1 hi
        asm volatile("tensor_load_to_lds %0, %1" :: "s"(g0), "s"(g1) : "memory");
        __builtin_amdgcn_s_wait_tensorcnt(0);
    }
    __syncthreads();

    // ---- batch max/min of v (closed-form softmax row max) ----
    float lmax = -3.402823466e38f, lmin = 3.402823466e38f;
    for (int idx = tid; idx < NN; idx += BLOCK) {
        const float val = sv[idx];
        lmax = fmaxf(lmax, val);
        lmin = fminf(lmin, val);
    }
    sred_max[tid] = lmax;
    sred_min[tid] = lmin;
    __syncthreads();
    for (int s = BLOCK / 2; s > 0; s >>= 1) {
        if (tid < s) {
            sred_max[tid] = fmaxf(sred_max[tid], sred_max[tid + s]);
            sred_min[tid] = fminf(sred_min[tid], sred_min[tid + s]);
        }
        __syncthreads();
    }
    const float vmax = sred_max[0];
    const float vmin = sred_min[0];

    // ---- per-wave 16-row i-tile; (s,t) reduction via v_wmma_f32_16x16x32_f16 ----
    const int wave = tid >> 5;
    const int lane = tid & 31;
    const int m    = lane & 15;     // A row / B column index
    const int hi   = lane >> 4;     // half-wave select
    const int i0   = tile * ROWS_PB + wave * 16;

    const float LOG2E = 1.44269504088896340736f;
    const float vi  = sv[i0 + m];
    const float vi2 = vi * LOG2E;
    const float mi2 = ((vi >= 0.0f) ? vi * vmax : vi * vmin) * LOG2E;
    const int hi8   = hi << 3;      // A-matrix K base per half-wave
    const int hi16  = hi << 4;      // unused; B also takes its K base from hi8

    // B (32x16 f16): col 0 = ones -> s, col 1 = v_j -> t, rest 0
    v16h bm;
#pragma unroll
    for (int kk = 0; kk < 16; ++kk)
        bm[kk] = (_Float16)((m == 0) ? 1.0f : 0.0f);

    v8f c = {};  // f32 accumulator: c[r] = (M=r or r+8, N=lane&15)

    auto step = [&](int j0) {
        // A (16x32 f16) K layout: lo half-wave K={0..7,16..23}, hi K={8..15,24..31}
        v16h a;
#pragma unroll
        for (int kk = 0; kk < 16; ++kk) {
            const int jA  = j0 + (((kk & 7) | ((kk >> 3) << 4)) + hi8);
            // hardware exp2 (v_exp_f32); result in (0,1] -> f16-safe
            const float e = __builtin_amdgcn_exp2f(fmaf(vi2, sv[jA], -mi2));
            a[kk] = (_Float16)e;
        }
        if (m == 1) {                 // only column-1 lanes refresh v_j chunk
#pragma unroll
            for (int kk = 0; kk < 16; ++kk)
                bm[kk] = (_Float16)sv[j0 + (((kk & 7) | ((kk >> 3) << 4)) + hi8)];
        }
        c = __builtin_amdgcn_wmma_f32_16x16x32_f16(
                false, a, false, bm, (short)0, c, false, false);
    };

    for (int j0 = 0; j0 < NN; j0 += 64) {   // 2 WMMA chunks per iteration
        step(j0);
        step(j0 + 32);
    }

    // ---- extract column 0 (s) and column 1 (t) from C layout via LDS ----
    if (m < 2) {                      // lanes 0,16 -> s ; lanes 1,17 -> t
        float* dst = (m == 0) ? sS : sT;
#pragma unroll
        for (int r = 0; r < 8; ++r)
            dst[wave * 16 + hi * 8 + r] = c[r];
    }
    __syncthreads();

    if (lane < 16) {
        const int row = wave * 16 + lane;
        const int i   = i0 + lane;
        out[b * NN + i] = gamma[0] * (sT[row] / sS[row]) + sv[i];
    }
}

extern "C" void kernel_launch(void* const* d_in, const int* in_sizes, int n_in,
                              void* d_out, int out_size, void* d_ws, size_t ws_size,
                              hipStream_t stream)
{
    (void)in_sizes; (void)n_in; (void)out_size; (void)d_ws; (void)ws_size;
    const float* x     = (const float*)d_in[0];  // [B, C, H, W] f32
    const float* gamma = (const float*)d_in[1];  // [1] f32
    float*       out   = (float*)d_out;          // [B, C, H, W] f32

    const int grid = NBATCH * (NN / ROWS_PB);    // 256 blocks
    gam_rank1_attn<<<grid, BLOCK, 0, stream>>>(x, gamma, out);
}
